// AE_90606630076419
// MI455X (gfx1250) — matrix operand order repaired and checked by reading, not compiled, not run on hardware
//
#include <hip/hip_runtime.h>
#include <math.h>

typedef __bf16 bf16;
typedef __attribute__((ext_vector_type(16))) __bf16 v16bf;
typedef __attribute__((ext_vector_type(8)))  __bf16 v8bf;
typedef __attribute__((ext_vector_type(8)))  float  v8f;

#define NFD   256          // feature dim
#define HF    128          // nf/2
#define LLEAF 131072
#define NNODES 262143
#define NLEV  17
#define KENC  160          // [X(5) pad->32 | h(128)]
#define KDEC  544          // [Pp(5) pad->32 | f(256) | h0(256)]

__device__ __forceinline__ bf16 f2bf(float f) {
  union { float f; unsigned u; } v; v.f = f;
  unsigned r = v.u + 0x7FFFu + ((v.u >> 16) & 1u);
  unsigned short h = (unsigned short)(r >> 16);
  union { unsigned short s; bf16 b; } o; o.s = h;
  return o.b;
}
__device__ __forceinline__ float sigm(float x) { return 1.0f / (1.0f + expf(-x)); }
__device__ __forceinline__ int offs(int k) { return 2 * LLEAF - ((2 * LLEAF) >> k); }

// Assemble one lane's 16-bf16 A operand from two contiguous 8-half chunks (contiguous k = 16h + s at the call sites, same order as B).
__device__ __forceinline__ v16bf ld_a(const bf16* row, int o1, int o2) {
  v8bf lo = *(const v8bf*)(row + o1);
  v8bf hi = *(const v8bf*)(row + o2);
  return __builtin_shufflevector(lo, hi, 0,1,2,3,4,5,6,7,8,9,10,11,12,13,14,15);
}

// ---------------------------------------------------------------------------
// Prep: pack weights to bf16 (K-padded), copy X -> X_ab, zero loss slots.
// ---------------------------------------------------------------------------
__global__ void prep_kernel(const float* __restrict__ X,
                            const float* __restrict__ eWih, const float* __restrict__ eWhh,
                            const float* __restrict__ fchW,
                            const float* __restrict__ dWih, const float* __restrict__ dWhh,
                            bf16* __restrict__ WencC, bf16* __restrict__ Wfch,
                            bf16* __restrict__ WdecC,
                            float* __restrict__ Xab, float* __restrict__ lossbuf) {
  const int T0 = 512 * KENC, T1 = 512 * NFD, T2 = 1024 * KDEC, T3 = NNODES * 5, T4 = 64;
  const int total = T0 + T1 + T2 + T3 + T4;
  for (int idx = blockIdx.x * blockDim.x + threadIdx.x; idx < total;
       idx += gridDim.x * blockDim.x) {
    int i = idx;
    if (i < T0) {
      int n = i / KENC, k = i % KENC;
      float v = (k < 5) ? eWih[n * 5 + k] : (k < 32 ? 0.0f : eWhh[n * HF + (k - 32)]);
      WencC[i] = f2bf(v);
    } else if ((i -= T0) < T1) {
      Wfch[i] = f2bf(fchW[i]);
    } else if ((i -= T1) < T2) {
      int n = i / KDEC, k = i % KDEC;
      float v;
      if (k < 5)        v = dWih[n * 261 + k];
      else if (k < 32)  v = 0.0f;
      else if (k < 288) v = dWih[n * 261 + 5 + (k - 32)];
      else              v = dWhh[n * NFD + (k - 288)];
      WdecC[i] = f2bf(v);
    } else if ((i -= T2) < T3) {
      Xab[i] = X[i];
    } else {
      lossbuf[i - T3] = 0.0f;
    }
  }
}

// ---------------------------------------------------------------------------
// Encoder level: 16 pairs per block. G(32x512) = [X|h]·WencCᵀ via bf16 WMMA.
// Wave w: m-tile = w&1 (left/right), j-tiles {2(w>>1),2(w>>1)+1} x 4 gates.
// ---------------------------------------------------------------------------
__global__ void __launch_bounds__(256)
enc_level_kernel(const float* __restrict__ X, const float* __restrict__ Feat0,
                 float* __restrict__ F, const bf16* __restrict__ WencC,
                 const float* __restrict__ bih, const float* __restrict__ bhh,
                 int level, int P) {
  __shared__ __align__(16) bf16 bfA[2][16][KENC];
  __shared__ float ph[16][HF];
  __shared__ float pc[16][HF];

  const int tid = threadIdx.x;
  const int pairBase = blockIdx.x * 16;
  const int childBase = offs(level);
  const int parentBase = offs(level + 1);

  for (int idx = tid; idx < 2 * 16 * KENC; idx += 256) {
    int t = idx / (16 * KENC);
    int rm = idx % (16 * KENC);
    int m = rm / KENC, k = rm % KENC;
    int pair = pairBase + m;
    float v = 0.0f;
    if (pair < P) {
      int node = childBase + 2 * pair + t;
      if (k < 5) v = X[node * 5 + k];
      else if (k >= 32)
        v = (level == 0) ? Feat0[node * NFD + (k - 32)]
                         : F[(size_t)(node - LLEAF) * NFD + (k - 32)];
    }
    bfA[t][m][k] = f2bf(v);
  }
  __syncthreads();

  const int w = tid >> 5, lane = tid & 31;
  const int laneN = lane & 15, hi = lane >> 4;
  const int mt = w & 1;
  const int jt0 = (w >> 1) * 2;

  const v8f vzero = {0.f, 0.f, 0.f, 0.f, 0.f, 0.f, 0.f, 0.f};
  v8f acc[2][4];
  for (int jt = 0; jt < 2; ++jt) for (int g = 0; g < 4; ++g) acc[jt][g] = vzero;

  for (int ks = 0; ks < KENC / 32; ++ks) {
    const bf16* arow = &bfA[mt][laneN][0];
    int kb = ks * 32 + hi * 16;
    v16bf a = ld_a(arow, kb, kb + 8);
    for (int jt = 0; jt < 2; ++jt)
      for (int g = 0; g < 4; ++g) {
        int N = (g * 8 + jt0 + jt) * 16 + laneN;
        v16bf b = *(const v16bf*)(WencC + (size_t)N * KENC + ks * 32 + hi * 16);
        acc[jt][g] = __builtin_amdgcn_wmma_f32_16x16x32_bf16(
            false, a, false, b, (short)0, acc[jt][g], false, false);
      }
  }

  float h2v[2][8], c2v[2][8];
  for (int jt = 0; jt < 2; ++jt) {
    int j = (jt0 + jt) * 16 + laneN;
    for (int e = 0; e < 8; ++e) {
      int m = e + hi * 8;
      int pair = pairBase + m;
      float gi = acc[jt][0][e] + bih[j] + bhh[j];
      float gf = acc[jt][1][e] + bih[HF + j] + bhh[HF + j];
      float gg = acc[jt][2][e] + bih[2 * HF + j] + bhh[2 * HF + j];
      float go = acc[jt][3][e] + bih[3 * HF + j] + bhh[3 * HF + j];
      float cin = 0.0f;
      if (pair < P) {
        int node = childBase + 2 * pair + mt;
        cin = (level == 0) ? Feat0[node * NFD + HF + j]
                           : F[(size_t)(node - LLEAF) * NFD + HF + j];
      }
      float c2 = sigm(gf) * cin + sigm(gi) * tanhf(gg);
      float h2 = sigm(go) * tanhf(c2);
      c2v[jt][e] = c2; h2v[jt][e] = h2;
    }
  }
  if (mt == 0) {
    for (int jt = 0; jt < 2; ++jt) {
      int j = (jt0 + jt) * 16 + laneN;
      for (int e = 0; e < 8; ++e) { int m = e + hi * 8; ph[m][j] = h2v[jt][e]; pc[m][j] = c2v[jt][e]; }
    }
  }
  __syncthreads();
  if (mt == 1) {
    for (int jt = 0; jt < 2; ++jt) {
      int j = (jt0 + jt) * 16 + laneN;
      for (int e = 0; e < 8; ++e) {
        int m = e + hi * 8;
        int pair = pairBase + m;
        if (pair < P) {
          float* dst = F + (size_t)(parentBase + pair - LLEAF) * NFD;
          dst[j] = h2v[jt][e] + ph[m][j];
          dst[HF + j] = c2v[jt][e] + pc[m][j];
        }
      }
    }
  }
}

// ---------------------------------------------------------------------------
// Decoder level: 16 parents per block.
// Phase A: z = relu(f·fc_hᵀ+b) (K=256); h0 half -> bf16 A-staging, c0 half -> LDS.
// Phase B: G2(16x1024) = [Pp|f|h0]·WdecCᵀ (K=544); in-register gate combine.
// Phase C: 6x128 heads, loss, X_ab box cascade.
// ---------------------------------------------------------------------------
__global__ void __launch_bounds__(256)
dec_level_kernel(float* __restrict__ Xab, float* __restrict__ F,
                 const bf16* __restrict__ Wfch, const float* __restrict__ fchB,
                 const bf16* __restrict__ WdecC,
                 const float* __restrict__ dbih, const float* __restrict__ dbhh,
                 const float* __restrict__ flW, const float* __restrict__ flB,
                 const float* __restrict__ frW, const float* __restrict__ frB,
                 float* __restrict__ lossbuf, int level, int P) {
  __shared__ __align__(16) bf16 bfA2[16][KDEC];
  __shared__ float c0buf[16][NFD];
  __shared__ float hbuf[16][NFD];
  __shared__ float ybuf[16][12];

  const int tid = threadIdx.x;
  const int rowBase = blockIdx.x * 16;
  const int childBase = offs(level);
  const int parentBase = offs(level + 1);

  for (int idx = tid; idx < 16 * 288; idx += 256) {
    int m = idx / 288, k = idx % 288;
    int row = rowBase + m;
    float v = 0.0f;
    if (row < P) {
      int p = parentBase + row;
      if (k < 5) v = Xab[p * 5 + k];
      else if (k >= 32) v = F[(size_t)(p - LLEAF) * NFD + (k - 32)];
    }
    bfA2[m][k] = f2bf(v);
  }
  __syncthreads();

  const int w = tid >> 5, lane = tid & 31;
  const int laneN = lane & 15, hi = lane >> 4;
  const v8f vzero = {0.f, 0.f, 0.f, 0.f, 0.f, 0.f, 0.f, 0.f};

  { // Phase A
    v8f acc[4];
    for (int t = 0; t < 4; ++t) acc[t] = vzero;
    for (int ks = 0; ks < 8; ++ks) {
      const bf16* arow = &bfA2[laneN][0];
      int kb = 32 + ks * 32 + hi * 16;
      v16bf a = ld_a(arow, kb, kb + 8);
      for (int t = 0; t < 4; ++t) {
        int N = (w * 4 + t) * 16 + laneN;
        v16bf b = *(const v16bf*)(Wfch + (size_t)N * NFD + ks * 32 + hi * 16);
        acc[t] = __builtin_amdgcn_wmma_f32_16x16x32_bf16(
            false, a, false, b, (short)0, acc[t], false, false);
      }
    }
    for (int t = 0; t < 4; ++t) {
      int col = (w * 4 + t) * 16 + laneN;
      for (int e = 0; e < 8; ++e) {
        int m = e + hi * 8;
        float z = fmaxf(acc[t][e] + fchB[col], 0.0f);
        if (col < NFD) bfA2[m][288 + col] = f2bf(z);
        else           c0buf[m][col - NFD] = z;
      }
    }
  }
  __syncthreads();

  { // Phase B
    v8f acc[2][4];
    for (int jt = 0; jt < 2; ++jt) for (int g = 0; g < 4; ++g) acc[jt][g] = vzero;
    for (int ks = 0; ks < KDEC / 32; ++ks) {
      const bf16* arow = &bfA2[laneN][0];
      int kb = ks * 32 + hi * 16;
      v16bf a = ld_a(arow, kb, kb + 8);
      for (int jt = 0; jt < 2; ++jt) {
        int jtt = w * 2 + jt;
        for (int g = 0; g < 4; ++g) {
          int N = g * NFD + jtt * 16 + laneN;
          v16bf b = *(const v16bf*)(WdecC + (size_t)N * KDEC + ks * 32 + hi * 16);
          acc[jt][g] = __builtin_amdgcn_wmma_f32_16x16x32_bf16(
              false, a, false, b, (short)0, acc[jt][g], false, false);
        }
      }
    }
    for (int jt = 0; jt < 2; ++jt) {
      int j = (w * 2 + jt) * 16 + laneN;   // 0..255
      for (int e = 0; e < 8; ++e) {
        int m = e + hi * 8;
        int row = rowBase + m;
        float gi = acc[jt][0][e] + dbih[j] + dbhh[j];
        float gf = acc[jt][1][e] + dbih[NFD + j] + dbhh[NFD + j];
        float gg = acc[jt][2][e] + dbih[2 * NFD + j] + dbhh[2 * NFD + j];
        float go = acc[jt][3][e] + dbih[3 * NFD + j] + dbhh[3 * NFD + j];
        float c0 = c0buf[m][j];
        float c2 = sigm(gf) * c0 + sigm(gi) * tanhf(gg);
        float ho = sigm(go) * tanhf(c2);
        hbuf[m][j] = ho;
        if (row < P) {
          int child = childBase + 2 * row + (j < HF ? 0 : 1);
          int jj = (j < HF) ? j : j - HF;
          if (child >= LLEAF) {  // leaf writes are dead: skip
            float* dst = F + (size_t)(child - LLEAF) * NFD;
            dst[jj] = ho;
            dst[HF + jj] = c2;
          }
        }
      }
    }
  }
  __syncthreads();

  // Phase C: yl/yr heads
  if (tid < 192) {
    int rrow = tid / 12;
    int s = (tid % 12) / 6;
    int o = tid % 6;
    const float* Wm = s ? frW : flW;
    const float* Bm = s ? frB : flB;
    float y = Bm[o];
    const float* hv = &hbuf[rrow][s * HF];
    for (int k = 0; k < HF; ++k) y += hv[k] * Wm[o * HF + k];
    ybuf[rrow][s * 6 + o] = y;
  }
  __syncthreads();
  if (tid < 32) {
    int rrow = tid >> 1, s = tid & 1;
    int row = rowBase + rrow;
    if (row < P) {
      int p = parentBase + row;
      int child = childBase + 2 * row + s;
      float y0 = ybuf[rrow][s * 6 + 0], y1 = ybuf[rrow][s * 6 + 1];
      float y2 = ybuf[rrow][s * 6 + 2], y3 = ybuf[rrow][s * 6 + 3];
      float y4 = ybuf[rrow][s * 6 + 4];
      float p0 = tanhf(y0), p1 = tanhf(y1), p2 = sigm(y2), p3 = sigm(y3), p4 = y4;
      float xc0 = Xab[child * 5 + 0], xc1 = Xab[child * 5 + 1], xc2 = Xab[child * 5 + 2];
      float xc3 = Xab[child * 5 + 3], xc4 = Xab[child * 5 + 4];
      float d0 = xc0 - p0, d1 = xc1 - p1, d2 = xc2 - p2, d3 = xc3 - p3, d4 = xc4 - p4;
      float err = d0 * d0 + d1 * d1 + d2 * d2 + d3 * d3 + d4 * d4;
      atomicAdd(&lossbuf[level], err / (5.0f * (float)P));
      float q0 = Xab[p * 5 + 0], q1 = Xab[p * 5 + 1], q2 = Xab[p * 5 + 2];
      float q3 = Xab[p * 5 + 3], q4 = Xab[p * 5 + 4];
      Xab[child * 5 + 0] = xc0 * q2 + q0;
      Xab[child * 5 + 1] = xc1 * q3 + q1;
      Xab[child * 5 + 2] = xc2 * q2;
      Xab[child * 5 + 3] = xc3 * q3;
      Xab[child * 5 + 4] = xc4 + q4;
    }
  }
}

__global__ void finish_kernel(const float* __restrict__ lossbuf, float* __restrict__ out) {
  if (threadIdx.x == 0 && blockIdx.x == 0) {
    float s = 0.0f;
    for (int k = 0; k < NLEV; ++k) s += lossbuf[k];
    out[0] = s / (float)NLEV;
  }
}

extern "C" void kernel_launch(void* const* d_in, const int* in_sizes, int n_in,
                              void* d_out, int out_size, void* d_ws, size_t ws_size,
                              hipStream_t stream) {
  (void)in_sizes; (void)n_in; (void)out_size; (void)ws_size;
  const float* X    = (const float*)d_in[0];
  const float* Ft0  = (const float*)d_in[1];
  const float* eWih = (const float*)d_in[2];
  const float* eWhh = (const float*)d_in[3];
  const float* ebih = (const float*)d_in[4];
  const float* ebhh = (const float*)d_in[5];
  const float* fchW = (const float*)d_in[6];
  const float* fchB = (const float*)d_in[7];
  const float* dWih = (const float*)d_in[8];
  const float* dWhh = (const float*)d_in[9];
  const float* dbih = (const float*)d_in[10];
  const float* dbhh = (const float*)d_in[11];
  const float* flW  = (const float*)d_in[12];
  const float* flB  = (const float*)d_in[13];
  const float* frW  = (const float*)d_in[14];
  const float* frB  = (const float*)d_in[15];

  char* ws = (char*)d_ws;
  size_t off = 0;
  auto alloc = [&](size_t bytes) {
    void* p = ws + off;
    off = (off + bytes + 255) & ~(size_t)255;
    return p;
  };
  float* F     = (float*)alloc((size_t)(NNODES - LLEAF) * NFD * sizeof(float)); // internal nodes only
  float* Xab   = (float*)alloc((size_t)NNODES * 5 * sizeof(float));
  bf16*  WencC = (bf16*)alloc((size_t)512 * KENC * sizeof(bf16));
  bf16*  WfchC = (bf16*)alloc((size_t)512 * NFD * sizeof(bf16));
  bf16*  WdecC = (bf16*)alloc((size_t)1024 * KDEC * sizeof(bf16));
  float* lossb = (float*)alloc(64 * sizeof(float));

  prep_kernel<<<2048, 256, 0, stream>>>(X, eWih, eWhh, fchW, dWih, dWhh,
                                        WencC, WfchC, WdecC, Xab, lossb);

  for (int k = 0; k < NLEV; ++k) {
    int P = (LLEAF >> k) / 2;             // pairs at this level
    int blocks = (P + 15) / 16;
    enc_level_kernel<<<blocks, 256, 0, stream>>>(X, Ft0, F, WencC, ebih, ebhh, k, P);
  }
  for (int k = NLEV - 1; k >= 0; --k) {
    int P = (LLEAF >> k) / 2;
    int blocks = (P + 15) / 16;
    dec_level_kernel<<<blocks, 256, 0, stream>>>(Xab, F, WfchC, fchB, WdecC, dbih, dbhh,
                                                 flW, flB, frW, frB, lossb, k, P);
  }
  finish_kernel<<<1, 1, 0, stream>>>(lossb, (float*)d_out);
}
